// GraphMambaNodeClassifier_31507880083601
// MI455X (gfx1250) — matrix operand order repaired and checked by reading, not compiled, not run on hardware
//
#include <hip/hip_runtime.h>
#include <hip/hip_bf16.h>
#include <math.h>

// ---------------------------------------------------------------------------
// GraphMamba node classifier, fused per-node WMMA pipeline for gfx1250.
//
// Sizes: N=16384, IN=256, HID=128, MAXL=128, NCLS=64, LAYERS=3, K=3.
// Strategy: keep each node's 128x128 (L x HID) activation tile resident in
// LDS (fp16), run every matmul through v_wmma_f32_16x16x32_f16, never spill
// the 1 GB "seq" tensor to HBM. Weight A-fragments are hoisted into VGPRs
// once per GEMM; B-fragments stream from LDS.
// ---------------------------------------------------------------------------

typedef __attribute__((ext_vector_type(16))) _Float16 v16h;
typedef __attribute__((ext_vector_type(8)))  _Float16 v8h;
typedef __attribute__((ext_vector_type(8)))  float    v8f;

#define NNODES 16384
#define INF    256
#define HID    128
#define MAXL   128
#define NCLS   64

// Workspace layout (in _Float16 elements):
//   [0]       localWT  [128][256]   (j,k)  32768
//   [32768]   ssmWT    [128][128]   (j,c)  16384
//   [49152]   convW    [3][3][128][128] (layer,tap,cout,cin) 147456
//   [196608]  outWT    [128][128]   (j,c)  16384
//   [212992]  H        [16384][128] fp16 encoded node features
#define WS_LOCALWT 0
#define WS_SSMWT   32768
#define WS_CONVW   49152
#define WS_OUTWT   196608
#define WS_H       212992

__device__ __forceinline__ float gelu_exact(float x) {
    return 0.5f * x * (1.0f + erff(x * 0.70710678118654752440f));
}

// A-fragment (16x32 f16, M x K), source row-major [M][ld], K-contiguous.
// Contiguous-k order used here (same as B frags): lanes 0-15 row m=lane, K = 0..15; lanes 16-31 same m,
// K = 16..31.  -> two contiguous 16B chunks per lane.
__device__ __forceinline__ v16h frag_a_mk(const _Float16* base, int ld,
                                          int mBase, int kBase, int lane) {
    int m = mBase + (lane & 15);
    int h = lane >> 4;
    const _Float16* p = base + (size_t)m * ld + kBase + h * 16;
    v8h lo = *(const v8h*)(p);
    v8h hi = *(const v8h*)(p + 8);
    v16h r;
#pragma unroll
    for (int i = 0; i < 8; ++i) { r[i] = lo[i]; r[i + 8] = hi[i]; }
    return r;
}

// B-fragment (32x16 f16, K x N), source stored [N][ld] (K-contiguous rows).
// Contiguous-k order, same as A frags: lanes 0-15 hold K=0..15 for col n=lane; lanes 16-31 K=16..31.
__device__ __forceinline__ v16h frag_b_nk(const _Float16* base, int ld,
                                          int nBase, int kBase, int lane) {
    int n = nBase + (lane & 15);
    int h = lane >> 4;
    const _Float16* p = base + (size_t)n * ld + kBase + h * 16;
    v8h lo = *(const v8h*)(p);
    v8h hi = *(const v8h*)(p + 8);
    v16h r;
#pragma unroll
    for (int i = 0; i < 8; ++i) { r[i] = lo[i]; r[i + 8] = hi[i]; }
    return r;
}

// Shifted B-fragment for dilated conv taps: row l+off, zero outside [0,128).
__device__ __forceinline__ v16h frag_b_shift(const _Float16* base,
                                             int nBase, int kBase, int off,
                                             int lane) {
    int n = nBase + (lane & 15) + off;
    v16h r = {};
    if (n >= 0 && n < MAXL) {
        int h = lane >> 4;
        const _Float16* p = base + (size_t)n * HID + kBase + h * 16;
        v8h lo = *(const v8h*)(p);
        v8h hi = *(const v8h*)(p + 8);
#pragma unroll
        for (int i = 0; i < 8; ++i) { r[i] = lo[i]; r[i + 8] = hi[i]; }
    }
    return r;
}

// B-fragment converted from fp32 row-major [N][ld] global memory.
__device__ __forceinline__ v16h frag_b_nk_f32(const float* base, int ld,
                                              int nBase, int kBase, int lane) {
    int n = nBase + (lane & 15);
    int h = lane >> 4;
    const float* p = base + (size_t)n * ld + kBase + h * 16;
    v16h r;
#pragma unroll
    for (int i = 0; i < 16; ++i) r[i] = (_Float16)p[i];
    return r;
}

// Store a D tile (M=out-channel rows, N=l cols) into buf[l][out] fp16 with
// per-row bias and optional exact GELU. Each lane's 8 rows are contiguous.
__device__ __forceinline__ void store_tile(_Float16* buf, v8f acc,
                                           int mBase, int nBase,
                                           const float* bias, bool doGelu,
                                           int lane) {
    int n  = nBase + (lane & 15);
    int m2 = mBase + (lane >> 4) * 8;
    v8h o;
#pragma unroll
    for (int v = 0; v < 8; ++v) {
        float x = acc[v] + bias[m2 + v];
        if (doGelu) x = gelu_exact(x);
        o[v] = (_Float16)x;
    }
    *(v8h*)(buf + (size_t)n * HID + m2) = o;
}

// ---------------------------------------------------------------------------
// Kernel 1: repack all weight matrices to fp16 in WMMA-friendly layouts.
// ---------------------------------------------------------------------------
__global__ void prep_weights_kernel(const float* __restrict__ localW,
                                    const float* __restrict__ ssmW,
                                    const float* __restrict__ convw,
                                    const float* __restrict__ outW,
                                    _Float16* __restrict__ ws) {
    int i = blockIdx.x * blockDim.x + threadIdx.x;
    int stride = gridDim.x * blockDim.x;
    for (int t = i; t < HID * INF; t += stride) {          // localWT [j][k]
        int j = t / INF, k = t % INF;
        ws[WS_LOCALWT + t] = (_Float16)localW[k * HID + j];
    }
    for (int t = i; t < HID * HID; t += stride) {          // ssmWT [j][c]
        int j = t / HID, c = t % HID;
        ws[WS_SSMWT + t] = (_Float16)ssmW[c * HID + j];
    }
    for (int t = i; t < 9 * HID * HID; t += stride) {      // convW [l][k][co][ci]
        int cin  = t % HID;
        int cout = (t / HID) % HID;
        int ik   = t / (HID * HID);
        int layer = ik / 3, k = ik % 3;
        ws[WS_CONVW + t] =
            (_Float16)convw[((size_t)(layer * HID + cout) * HID + cin) * 3 + k];
    }
    for (int t = i; t < HID * HID; t += stride) {          // outWT [j][c]
        int j = t / HID, c = t % HID;
        ws[WS_OUTWT + t] = (_Float16)outW[c * HID + j];
    }
}

// ---------------------------------------------------------------------------
// Kernel 2: H = LN(GELU(X @ local_W + b)) -> fp16 workspace. 64 nodes/block.
// WMMA with M=hidden(128, 8 tiles over 8 waves), N=node(64, 4 tiles), K=256.
// ---------------------------------------------------------------------------
__global__ __launch_bounds__(256)
void encode_kernel(const float* __restrict__ X,
                   const _Float16* __restrict__ ws,
                   const float* __restrict__ lb,
                   const float* __restrict__ lnw,
                   const float* __restrict__ lnb,
                   _Float16* __restrict__ H) {
    __shared__ _Float16 enc[64 * HID];          // [node][j] fp16, 16 KB
    __shared__ float esum[256], esq[256];
    __shared__ float emean[64], erstd[64];
    const int tid  = threadIdx.x;
    const int lane = tid & 31;
    const int wave = tid >> 5;
    const int nodeBase = blockIdx.x * 64;
    const _Float16* localWT = ws + WS_LOCALWT;

    const int mBase = wave * 16;
    // hoist loop-invariant weight A-fragments (K=256 -> 8 fragments)
    v16h aW[8];
#pragma unroll
    for (int kk = 0; kk < 8; ++kk)
        aW[kk] = frag_a_mk(localWT, INF, mBase, kk * 32, lane);

#pragma unroll
    for (int nt = 0; nt < 4; ++nt) {
        v8f acc = {};
#pragma unroll
        for (int kk = 0; kk < 8; ++kk) {
            v16h b = frag_b_nk_f32(X + (size_t)nodeBase * INF, INF,
                                   nt * 16, kk * 32, lane);
            acc = __builtin_amdgcn_wmma_f32_16x16x32_f16(
                false, aW[kk], false, b, (short)0, acc, false, false);
        }
        int n  = nt * 16 + (lane & 15);
        int m2 = mBase + (lane >> 4) * 8;
        v8h o;
#pragma unroll
        for (int v = 0; v < 8; ++v)
            o[v] = (_Float16)gelu_exact(acc[v] + lb[m2 + v]);
        *(v8h*)(enc + (size_t)n * HID + m2) = o;
    }
    __syncthreads();

    // two-phase LayerNorm over hidden axis: 4 threads per node row
    {
        int node = tid >> 2, q = tid & 3;
        float s = 0.f, ss = 0.f;
        for (int j = q * 32; j < q * 32 + 32; ++j) {
            float x = (float)enc[node * HID + j];
            s += x; ss += x * x;
        }
        esum[tid] = s; esq[tid] = ss;
    }
    __syncthreads();
    if (tid < 64) {
        float s  = esum[tid*4] + esum[tid*4+1] + esum[tid*4+2] + esum[tid*4+3];
        float ss = esq[tid*4]  + esq[tid*4+1]  + esq[tid*4+2]  + esq[tid*4+3];
        float m   = s * (1.0f / HID);
        float var = ss * (1.0f / HID) - m * m;
        emean[tid] = m;
        erstd[tid] = rsqrtf(var + 1e-5f);
    }
    __syncthreads();
    {
        int node = tid >> 2, q = tid & 3;
        float m = emean[node], rs = erstd[node];
        _Float16* dst = H + (size_t)(nodeBase + node) * HID;
        for (int j = q * 32; j < q * 32 + 32; ++j) {
            float x = ((float)enc[node * HID + j] - m) * rs * lnw[j] + lnb[j];
            dst[j] = (_Float16)x;
        }
    }
}

// ---------------------------------------------------------------------------
// Kernel 3: fused per-node pipeline. One node per 256-thread block (8 waves).
// ---------------------------------------------------------------------------
__global__ __launch_bounds__(256)
void node_pipeline_kernel(const _Float16* __restrict__ ws,
                          const int* __restrict__ nbr,
                          const int* __restrict__ nlen,
                          const float* __restrict__ ssm_b,
                          const float* __restrict__ conv_b,
                          const float* __restrict__ lnw,
                          const float* __restrict__ lnb,
                          const float* __restrict__ out_b,
                          const float* __restrict__ W1,
                          const float* __restrict__ b1,
                          const float* __restrict__ W2,
                          const float* __restrict__ b2,
                          float* __restrict__ out) {
    extern __shared__ char smem[];
    _Float16* bufA   = (_Float16*)smem;                 // [l][c] 32 KB
    _Float16* bufB   = bufA + MAXL * HID;               // [l][c] 32 KB
    float*    pooled = (float*)(bufB + MAXL * HID);     // [128]
    float*    z1     = pooled + HID;                    // [64]
    float*    lnsum  = z1 + NCLS;                       // [256]
    float*    lnsq   = lnsum + 256;                     // [256]
    float*    aArr   = lnsq + 256;                      // [128]
    float*    bArr   = aArr + HID;                      // [128]

    const int n    = blockIdx.x;
    const int tid  = threadIdx.x;
    const int lane = tid & 31;
    const int wave = tid >> 5;

    const _Float16* ssmWT = ws + WS_SSMWT;
    const _Float16* convW = ws + WS_CONVW;
    const _Float16* outWT = ws + WS_OUTWT;
    const _Float16* H     = ws + WS_H;

    // warm L2/WGP$ with the (device-wide shared) weight matrices
    __builtin_prefetch(ssmWT + (size_t)tid * 64, 0, 0);
    __builtin_prefetch(convW + (size_t)tid * 576, 0, 0);
    __builtin_prefetch(outWT + (size_t)tid * 64, 0, 0);

    const int len = nlen[n];

    // --- 1. masked ragged gather of neighbor rows: bufA[l][c] = H[idx[l]] ---
    {
        int l  = tid >> 1;
        int hh = tid & 1;
        int idx   = nbr[n * MAXL + l];
        bool valid = l < len;
        const v8h* src = (const v8h*)(H + (size_t)idx * HID + hh * 64);
        v8h* dst = (v8h*)(bufA + (size_t)l * HID + hh * 64);
#pragma unroll
        for (int t = 0; t < 8; ++t) {
            v8h z = {};
            dst[t] = valid ? src[t] : z;
        }
    }
    if (tid < HID) pooled[tid] = 0.f;
    __syncthreads();

    const int mBase = wave * 16;

    // --- 2. GEMM1: bufB[l][j] = seq @ ssm_in_W + b  (M=j, N=l, K=c) ---
    {
        v16h aW[4];
#pragma unroll
        for (int kk = 0; kk < 4; ++kk)
            aW[kk] = frag_a_mk(ssmWT, HID, mBase, kk * 32, lane);
#pragma unroll
        for (int nt = 0; nt < 8; ++nt) {
            v8f acc = {};
#pragma unroll
            for (int kk = 0; kk < 4; ++kk) {
                v16h b = frag_b_nk(bufA, HID, nt * 16, kk * 32, lane);
                acc = __builtin_amdgcn_wmma_f32_16x16x32_f16(
                    false, aW[kk], false, b, (short)0, acc, false, false);
            }
            store_tile(bufB, acc, mBase, nt * 16, ssm_b, false, lane);
        }
    }

    // --- 3. three dilated conv layers: 3 taps x WMMA, bias+GELU, LN over l ---
    _Float16* cur = bufB;
    _Float16* nxt = bufA;
#pragma unroll
    for (int layer = 0; layer < 3; ++layer) {
        __syncthreads();
        const int d = 1 << layer;
        // hoist the 12 weight A-fragments (3 taps x 4 k-steps) for this layer
        v16h aW[3][4];
#pragma unroll
        for (int k = 0; k < 3; ++k)
#pragma unroll
            for (int kk = 0; kk < 4; ++kk)
                aW[k][kk] = frag_a_mk(convW + (size_t)(layer * 3 + k) * HID * HID,
                                      HID, mBase, kk * 32, lane);
#pragma unroll
        for (int nt = 0; nt < 8; ++nt) {
            v8f acc = {};
#pragma unroll
            for (int k = 0; k < 3; ++k) {
                const int off = (k - 1) * d;
#pragma unroll
                for (int kk = 0; kk < 4; ++kk) {
                    v16h b = frag_b_shift(cur, nt * 16, kk * 32, off, lane);
                    acc = __builtin_amdgcn_wmma_f32_16x16x32_f16(
                        false, aW[k][kk], false, b, (short)0, acc, false, false);
                }
            }
            store_tile(nxt, acc, mBase, nt * 16, conv_b + layer * HID, true, lane);
        }
        __syncthreads();

        // two-phase per-channel LayerNorm over L: 256 threads, half rows each
        {
            int c  = tid & 127;
            int hh = tid >> 7;
            float s = 0.f, q = 0.f;
            for (int l = hh * 64; l < hh * 64 + 64; ++l) {
                float x = (float)nxt[(size_t)l * HID + c];
                s += x; q += x * x;
            }
            lnsum[tid] = s; lnsq[tid] = q;
        }
        __syncthreads();
        if (tid < HID) {
            float s = lnsum[tid] + lnsum[tid + 128];
            float q = lnsq[tid] + lnsq[tid + 128];
            float m   = s * (1.0f / MAXL);
            float var = q * (1.0f / MAXL) - m * m;
            float rs  = rsqrtf(var + 1e-5f);
            float w = lnw[layer * HID + tid], bb = lnb[layer * HID + tid];
            float a = rs * w;
            aArr[tid] = a;
            bArr[tid] = bb - m * a;          // x' = x*a + (b - m*a)
        }
        __syncthreads();
        {
            int c  = tid & 127;
            int hh = tid >> 7;
            float a = aArr[c], b0 = bArr[c];
            for (int l = hh * 64; l < hh * 64 + 64; ++l) {
                float x = (float)nxt[(size_t)l * HID + c];
                nxt[(size_t)l * HID + c] = (_Float16)(x * a + b0);
            }
        }
        _Float16* t = cur; cur = nxt; nxt = t;
    }
    __syncthreads();

    // --- 4. out GEMM (M=j, N=l) + masked mean pooling via LDS atomics ---
    {
        v16h aW[4];
#pragma unroll
        for (int kk = 0; kk < 4; ++kk)
            aW[kk] = frag_a_mk(outWT, HID, mBase, kk * 32, lane);
        int m2 = mBase + (lane >> 4) * 8;
        float pp[8];
#pragma unroll
        for (int v = 0; v < 8; ++v) pp[v] = 0.f;
#pragma unroll
        for (int nt = 0; nt < 8; ++nt) {
            v8f acc = {};
#pragma unroll
            for (int kk = 0; kk < 4; ++kk) {
                v16h b = frag_b_nk(cur, HID, nt * 16, kk * 32, lane);
                acc = __builtin_amdgcn_wmma_f32_16x16x32_f16(
                    false, aW[kk], false, b, (short)0, acc, false, false);
            }
            int l = nt * 16 + (lane & 15);
            float valid = (l < len) ? 1.f : 0.f;
#pragma unroll
            for (int v = 0; v < 8; ++v)
                pp[v] += (acc[v] + out_b[m2 + v]) * valid;
        }
#pragma unroll
        for (int v = 0; v < 8; ++v) atomicAdd(&pooled[m2 + v], pp[v]);
    }
    __syncthreads();

    // --- 5. classifier head (tiny; plain VALU in fp32) ---
    const float denom = (len > 0) ? (float)len : 1.0f;
    if (tid < NCLS) {
        float s = b1[tid];
        for (int c = 0; c < HID; ++c)
            s += (pooled[c] / denom) * W1[c * NCLS + tid];
        z1[tid] = gelu_exact(s);
    }
    __syncthreads();
    if (tid < NCLS) {
        float s = b2[tid];
        for (int t = 0; t < NCLS; ++t) s += z1[t] * W2[t * NCLS + tid];
        out[(size_t)n * NCLS + tid] = s;
    }
}

// ---------------------------------------------------------------------------
extern "C" void kernel_launch(void* const* d_in, const int* in_sizes, int n_in,
                              void* d_out, int out_size, void* d_ws, size_t ws_size,
                              hipStream_t stream) {
    const float* node_feat  = (const float*)d_in[0];
    const int*   nbr        = (const int*)d_in[1];
    const int*   nlen       = (const int*)d_in[2];
    const float* localW     = (const float*)d_in[3];
    const float* local_b    = (const float*)d_in[4];
    const float* local_ln_w = (const float*)d_in[5];
    const float* local_ln_b = (const float*)d_in[6];
    const float* ssmW       = (const float*)d_in[7];
    const float* ssm_b      = (const float*)d_in[8];
    const float* convw      = (const float*)d_in[9];
    const float* conv_b     = (const float*)d_in[10];
    const float* lnw        = (const float*)d_in[11];
    const float* lnb        = (const float*)d_in[12];
    const float* outW       = (const float*)d_in[13];
    const float* out_b      = (const float*)d_in[14];
    const float* W1         = (const float*)d_in[15];
    const float* b1         = (const float*)d_in[16];
    const float* W2         = (const float*)d_in[17];
    const float* b2         = (const float*)d_in[18];
    float*     out = (float*)d_out;
    _Float16*  ws  = (_Float16*)d_ws;

    prep_weights_kernel<<<256, 256, 0, stream>>>(localW, ssmW, convw, outW, ws);

    encode_kernel<<<NNODES / 64, 256, 0, stream>>>(
        node_feat, ws, local_b, local_ln_w, local_ln_b, ws + WS_H);

    const size_t dynLds = (size_t)2 * MAXL * HID * sizeof(_Float16)  // bufA+bufB
                        + HID * sizeof(float)                        // pooled
                        + NCLS * sizeof(float)                       // z1
                        + 2 * 256 * sizeof(float)                    // lnsum/lnsq
                        + 2 * HID * sizeof(float);                   // aArr/bArr
    node_pipeline_kernel<<<NNODES, 256, dynLds, stream>>>(
        ws, nbr, nlen, ssm_b, conv_b, lnw, lnb, out_b, W1, b1, W2, b2, out);
}
